// Convolution_4724464026042
// MI455X (gfx1250) — matrix operand order repaired and checked by reading, not compiled, not run on hardware
//
#include <hip/hip_runtime.h>
#include <hip/hip_bf16.h>
#include <math.h>

typedef __attribute__((ext_vector_type(16))) _Float16 v16h;
typedef __attribute__((ext_vector_type(8)))  float    v8f;

#define MDIM 4
#define ADIM 256
#define RBF  64
#define FDIM 32
#define HDIM 32
#define SI   32
#define NCOL 128          // 4 filters * 32 channels
#define EPSV 1e-7f

__device__ __forceinline__ float ssp(float x) {
    // shifted softplus: log(0.5*exp(x)+0.5)
    return logf(0.5f * expf(x) + 0.5f);
}

// ---------------------------------------------------------------------------
// Kernel 1: collapse the two linear Dense layers of each radial MLP into one
// 64x128 weight, written DIRECTLY in the per-lane WMMA B-matrix swizzle:
//   element e = nt*1024 + ch*512 + lane*16 + idx   (f16)
//   holds Wc[k][col] with k = ch*32 + (lane>>4)*16 + idx, col = nt*16 + (lane&15)
// so the main kernel fetches each lane's v16h with one aligned 32B LDS read.
// Also writes the 128 fused f32 biases.
// ---------------------------------------------------------------------------
__global__ void prep_weights(const float* w1_00, const float* b1_00, const float* w2_00, const float* b2_00,
                             const float* w1_01, const float* b1_01, const float* w2_01, const float* b2_01,
                             const float* w1_10, const float* b1_10, const float* w2_10, const float* b2_10,
                             const float* w1_11, const float* b1_11, const float* w2_11, const float* b2_11,
                             _Float16* bswz, float* bc) {
    const float* W1[4] = {w1_00, w1_01, w1_10, w1_11};
    const float* B1[4] = {b1_00, b1_01, b1_10, b1_11};
    const float* W2[4] = {w2_00, w2_01, w2_10, w2_11};
    const float* B2[4] = {b2_00, b2_01, b2_10, b2_11};
    int tid = threadIdx.x;
    for (int e = tid; e < NCOL * RBF; e += blockDim.x) {
        int idx  = e & 15;
        int lane = (e >> 4) & 31;
        int ch   = (e >> 9) & 1;
        int nt   = e >> 10;
        int k    = ch * 32 + (lane >> 4) * 16 + idx;   // 0..63
        int col  = nt * 16 + (lane & 15);              // 0..127
        int t    = col >> 5;
        int f    = col & 31;
        const float* w1 = W1[t];
        const float* w2 = W2[t];
        float s = 0.f;
        #pragma unroll
        for (int h = 0; h < HDIM; ++h)
            s += w1[k * HDIM + h] * w2[h * FDIM + f];
        bswz[e] = (_Float16)s;
    }
    if (tid < NCOL) {
        int t = tid >> 5, f = tid & 31;
        const float* w2 = W2[t];
        float s = B2[t][f];
        #pragma unroll
        for (int h = 0; h < HDIM; ++h)
            s += B1[t][h] * w2[h * FDIM + f];
        bc[tid] = s;
    }
}

// ---------------------------------------------------------------------------
// Kernel 2: fused radial GEMM (WMMA f16->f32) + neighbor contraction +
// self-interaction + equivariant activation.
// Block = 8 waves of 32; wave w handles atom a = blockIdx.x*8 + w, m = blockIdx.y.
// Two passes over the neighbor loop (ntiles 0-3 then 4-7) keep VGPR use low.
// ---------------------------------------------------------------------------
__global__ void __launch_bounds__(256)
conv_kernel(const float* __restrict__ image,
            const float* __restrict__ vectors,
            const float* __restrict__ feat0,
            const float* __restrict__ feat1,
            const _Float16* __restrict__ bswz,
            const float* __restrict__ bc,
            const float* __restrict__ w_si0,
            const float* __restrict__ w_si1,
            const float* __restrict__ b_act0,
            const float* __restrict__ b_act1,
            float* __restrict__ out) {
    __shared__ _Float16 sB[NCOL * RBF];         // 16 KB pre-swizzled B
    __shared__ float    sVec[8][ADIM * 4];      // 32 KB per-wave: vx,vy,vz,maskflag
    __shared__ float    sCat[8][64 + 96 * 3];   // 11 KB per-wave cat0/cat1 scratch

    const int lane = threadIdx.x;               // 0..31 (wave32)
    const int w    = threadIdx.y;               // wave id 0..7
    const int m    = blockIdx.y;
    const int a    = blockIdx.x * 8 + w;
    const int tid  = w * 32 + lane;
    const int n    = lane & 15;
    const int kh   = lane >> 4;

    // cooperative 16B-vector copy of swizzled weights into LDS
    {
        const uint4* src = (const uint4*)bswz;
        uint4* dst = (uint4*)sB;
        for (int e = tid; e < (NCOL * RBF) / 8; e += 256) dst[e] = src[e];
    }

    // per-wave: vector row + mask flag for this (m,a)
    for (int b = lane; b < ADIM; b += 32) {
        const float* vp = vectors + (((size_t)m * ADIM + a) * ADIM + b) * 3;
        float vx = vp[0], vy = vp[1], vz = vp[2];
        float nrm = sqrtf(vx * vx + vy * vy + vz * vz);
        float flag = (nrm < EPSV) ? 0.f : 1.f;
        sVec[w][b * 4 + 0] = vx;
        sVec[w][b * 4 + 1] = vy;
        sVec[w][b * 4 + 2] = vz;
        sVec[w][b * 4 + 3] = flag;
    }
    __syncthreads();

    // per-lane fused biases for the 8 N-tiles (lane owns column f = nt*16+n)
    float bcf[8];
    #pragma unroll
    for (int nt = 0; nt < 8; ++nt) bcf[nt] = bc[nt * 16 + n];

    const v16h*  Bp        = (const v16h*)sB;   // index: nt*2 + ch, times 32 lanes
    const float* imrowbase = image + (((size_t)m * ADIM + a) * ADIM) * RBF;

    #pragma unroll
    for (int pass = 0; pass < 2; ++pass) {
        const int ntbase = pass * 4;

        // hoist 4 ntiles x 2 K-chunks of B: one aligned 32B LDS read each
        v16h Breg[8];
        #pragma unroll
        for (int q = 0; q < 4; ++q) {
            #pragma unroll
            for (int ch = 0; ch < 2; ++ch)
                Breg[q * 2 + ch] = Bp[((ntbase + q) * 2 + ch) * 32 + lane];
        }

        float accA[2]    = {0.f, 0.f};   // pass0: out_0x0_0 ; pass1: out_1x1_0
        float accV[2][3] = {};           // pass0: out_0x1_1 ; pass1: out_1x1_1
        float accW[2][3] = {};           // pass1 only: out_1x0_1

        for (int bt = 0; bt < 16; ++bt) {
            // A-matrix 16x32 f16 per K-chunk (contiguous k = 16*kh + s, same as B)
            const float* arow = imrowbase + (size_t)(bt * 16 + n) * RBF;
            v16h Areg[2];
            #pragma unroll
            for (int ch = 0; ch < 2; ++ch) {
                v16h av;
                int kb = ch * 32 + 16 * kh;
                #pragma unroll
                for (int j = 0; j < 4; ++j) {
                    float2 lo = *(const float2*)(arow + kb + 2 * j);
                    float2 hi = *(const float2*)(arow + kb + 8 + 2 * j);
                    av[2 * j]         = (_Float16)lo.x;
                    av[2 * j + 1]     = (_Float16)lo.y;
                    av[8 + 2 * j]     = (_Float16)hi.x;
                    av[8 + 2 * j + 1] = (_Float16)hi.y;
                }
                Areg[ch] = av;
            }

            // 4 radial tiles for this pass (bias folded into C)
            v8f c[4];
            #pragma unroll
            for (int q = 0; q < 4; ++q) {
                v8f ci;
                #pragma unroll
                for (int r = 0; r < 8; ++r) ci[r] = bcf[ntbase + q];
                ci = __builtin_amdgcn_wmma_f32_16x16x32_f16(false, Areg[0], false, Breg[q * 2],
                                                            (short)0, ci, false, false);
                ci = __builtin_amdgcn_wmma_f32_16x16x32_f16(false, Areg[1], false, Breg[q * 2 + 1],
                                                            (short)0, ci, false, false);
                c[q] = ci;
            }

            // joint consumption: ntile q and q+2 share channel f = q*16+n
            #pragma unroll
            for (int r = 0; r < 8; ++r) {
                int b = bt * 16 + 8 * kh + r;
                const float* vv = &sVec[w][b * 4];
                float vx = vv[0], vy = vv[1], vz = vv[2], fl = vv[3];
                if (pass == 0) {
                    #pragma unroll
                    for (int p = 0; p < 2; ++p) {
                        int f = p * 16 + n;
                        float f0 = feat0[((size_t)m * ADIM + b) * FDIM + f];
                        accA[p] += c[p][r] * f0;                 // out_0x0_0
                        float s = c[p + 2][r] * fl * f0;         // masked r01
                        accV[p][0] += s * vx;
                        accV[p][1] += s * vy;
                        accV[p][2] += s * vz;
                    }
                } else {
                    #pragma unroll
                    for (int p = 0; p < 2; ++p) {
                        int f = p * 16 + n;
                        const float* f1 = feat1 + (((size_t)m * ADIM + b) * FDIM + f) * 3;
                        float fx = f1[0], fy = f1[1], fz = f1[2];
                        float v10 = c[p][r];                     // r10
                        accW[p][0] += v10 * fx;
                        accW[p][1] += v10 * fy;
                        accW[p][2] += v10 * fz;
                        float s = c[p + 2][r] * fl;              // masked r11
                        accA[p]    += s * (vx * fx + vy * fy + vz * fz);   // 1x1->0
                        accV[p][0] += s * (vy * fz - vz * fy);             // 1x1->1
                        accV[p][1] += s * (vz * fx - vx * fz);
                        accV[p][2] += s * (vx * fy - vy * fx);
                    }
                }
            }
        }

        // fold half-wave partials (lanes L and L^16 own the same f)
        #pragma unroll
        for (int p = 0; p < 2; ++p) {
            accA[p] += __shfl_xor(accA[p], 16, 32);
            #pragma unroll
            for (int i = 0; i < 3; ++i) {
                accV[p][i] += __shfl_xor(accV[p][i], 16, 32);
                accW[p][i] += __shfl_xor(accW[p][i], 16, 32);
            }
        }

        // write this pass's channels into cat0[64] / cat1[96][3]
        if (lane < 16) {
            #pragma unroll
            for (int p = 0; p < 2; ++p) {
                int f = p * 16 + n;
                if (pass == 0) {
                    sCat[w][f] = accA[p];                                   // out_0x0_0
                    #pragma unroll
                    for (int i = 0; i < 3; ++i)
                        sCat[w][64 + f * 3 + i] = accV[p][i];               // out_0x1_1
                } else {
                    sCat[w][32 + f] = accA[p];                              // out_1x1_0
                    #pragma unroll
                    for (int i = 0; i < 3; ++i) {
                        sCat[w][64 + (32 + f) * 3 + i] = accW[p][i];        // out_1x0_1
                        sCat[w][64 + (64 + f) * 3 + i] = accV[p][i];        // out_1x1_1
                    }
                }
            }
        }
    }
    __syncthreads();

    // self-interaction + equivariant activation: lane g handles output channel g
    int g = lane;
    float si0 = 0.f;
    #pragma unroll 8
    for (int f = 0; f < 64; ++f) si0 += sCat[w][f] * w_si0[g * 64 + f];
    float o0 = ssp(si0 + b_act0[g]);
    out[((size_t)m * ADIM + a) * SI + g] = o0;

    float s1x = 0.f, s1y = 0.f, s1z = 0.f;
    #pragma unroll 8
    for (int f = 0; f < 96; ++f) {
        float wv = w_si1[g * 96 + f];
        s1x += sCat[w][64 + f * 3 + 0] * wv;
        s1y += sCat[w][64 + f * 3 + 1] * wv;
        s1z += sCat[w][64 + f * 3 + 2] * wv;
    }
    float ss = s1x * s1x + s1y * s1y + s1z * s1z;
    float n1 = sqrtf(fmaxf(ss, EPSV));
    float sc = ssp(n1 + b_act1[g]) / n1;
    size_t base1 = (size_t)MDIM * ADIM * SI + (((size_t)m * ADIM + a) * SI + g) * 3;
    out[base1 + 0] = s1x * sc;
    out[base1 + 1] = s1y * sc;
    out[base1 + 2] = s1z * sc;
}

extern "C" void kernel_launch(void* const* d_in, const int* in_sizes, int n_in,
                              void* d_out, int out_size, void* d_ws, size_t ws_size,
                              hipStream_t stream) {
    const float* image   = (const float*)d_in[0];
    const float* vectors = (const float*)d_in[1];
    const float* feat0   = (const float*)d_in[2];
    const float* feat1   = (const float*)d_in[3];
    const float* mp[16];
    for (int i = 0; i < 16; ++i) mp[i] = (const float*)d_in[4 + i];
    const float* w_si0  = (const float*)d_in[20];
    const float* w_si1  = (const float*)d_in[21];
    const float* b_act0 = (const float*)d_in[22];
    const float* b_act1 = (const float*)d_in[23];

    _Float16* bswz = (_Float16*)d_ws;                        // 8192 f16 = 16 KB
    float*    bcp  = (float*)((char*)d_ws + NCOL * RBF * 2); // 128 f32

    prep_weights<<<1, 256, 0, stream>>>(
        mp[0], mp[1], mp[2], mp[3],
        mp[4], mp[5], mp[6], mp[7],
        mp[8], mp[9], mp[10], mp[11],
        mp[12], mp[13], mp[14], mp[15],
        bswz, bcp);

    dim3 grid(ADIM / 8, MDIM);
    dim3 block(32, 8);
    conv_kernel<<<grid, block, 0, stream>>>(
        image, vectors, feat0, feat1, bswz, bcp,
        w_si0, w_si1, b_act0, b_act1, (float*)d_out);
}
